// AttentionDecoderGRU_33423435497664
// MI455X (gfx1250) — matrix operand order repaired and checked by reading, not compiled, not run on hardware
//
#include <hip/hip_runtime.h>
#include <hip/hip_bf16.h>

// ---------------------------------------------------------------------------
// AttentionDecoderGRU for MI455X (gfx1250, wave32, WMMA + async/TDM staging)
//
//  * Encoder K/V projections hoisted out of the 20-step loop (~11 TFLOP saved)
//  * Self-attn seq-len==1 -> softmax==1 -> only the V path survives
//  * All GEMMs: bf16 operands (pre-converted once), v_wmma_f32_16x16x32_bf16
//  * A tiles: GLOBAL_LOAD_ASYNC_TO_LDS_B128 (ASYNCcnt); W tiles: TDM
//    (tensor_load_to_lds, TENSORcnt); double-buffered so DMA overlaps WMMA.
// ---------------------------------------------------------------------------

#define H_DIM   1024
#define BATCH_N 1024
#define SEQ_N   128
#define TSTEPS  20

typedef unsigned short u16;
typedef __attribute__((ext_vector_type(16))) __bf16     v16bf;
typedef __attribute__((ext_vector_type(8)))  float      v8f;
typedef __attribute__((ext_vector_type(4)))  unsigned   v4u;
typedef __attribute__((ext_vector_type(8)))  unsigned   v8u;

__device__ __forceinline__ unsigned f2bf(float f) {
  unsigned u = __float_as_uint(f);
  u += 0x7FFFu + ((u >> 16) & 1u);   // round-to-nearest-even
  return u >> 16;
}

union FragU { uint4 q[2]; v16bf v; };

// ---- CDNA5 async / tensor staging primitives (inline asm: toolchain-portable)
__device__ __forceinline__ void async_ld_b128(unsigned lds_byte_off, const void* g) {
  asm volatile("global_load_async_to_lds_b128 %0, %1, off"
               :: "v"(lds_byte_off), "v"((unsigned long long)(size_t)g)
               : "memory");
}
__device__ __forceinline__ void wait_async0() {
#if __has_builtin(__builtin_amdgcn_s_wait_asynccnt)
  __builtin_amdgcn_s_wait_asynccnt(0);
#else
  asm volatile("s_wait_asynccnt 0x0" ::: "memory");
#endif
}
__device__ __forceinline__ void wait_tensor0() {
#if __has_builtin(__builtin_amdgcn_s_wait_tensorcnt)
  __builtin_amdgcn_s_wait_tensorcnt(0);
#else
  asm volatile("s_wait_tensorcnt 0x0" ::: "memory");
#endif
}

// TDM: load a 32(K) x 128(rows) bf16 tile, row stride K elements, into LDS with
// 16B padding every 64B so it lands in the LDT=40-ushort padded layout.
__device__ __forceinline__ void tdm_load_tile(unsigned lds_off, const u16* gsrc, int K) {
  unsigned long long ga = (unsigned long long)(size_t)gsrc;
  v4u g0;
  g0[0] = 1u;                                   // count=1, user descriptor
  g0[1] = lds_off;                              // lds_addr (bytes)
  g0[2] = (unsigned)ga;                         // global_addr[31:0]
  g0[3] = (unsigned)((ga >> 32) & 0x1FFFFFFu) | (2u << 30);  // addr[56:32], type=2
  v8u g1;
  g1[0] = (1u << 16)        // data_size = 2B
        | (1u << 20)        // pad_enable
        | (3u << 22)        // pad_interval: 16 DWORDs (64B) between pads
        | (3u << 25);       // pad_amount: 4 DWORDs (16B)
  g1[1] = ((unsigned)K & 0xFFFFu) << 16;                 // tensor_dim0[15:0]
  g1[2] = ((unsigned)K >> 16) | (128u << 16);            // tensor_dim0[31:16], tensor_dim1=128
  g1[3] = (32u << 16);                                   // tile_dim0 = 32
  g1[4] = 128u;                                          // tile_dim1 = 128, tile_dim2 = 0
  g1[5] = (unsigned)K;                                   // tensor_dim0_stride[31:0]
  g1[6] = 0u;
  g1[7] = 0u;
  asm volatile("tensor_load_to_lds %0, %1" :: "s"(g0), "s"(g1) : "memory");
}

// ---------------------------------------------------------------------------
// GEMM: C[M,N] = A[M,K](bf16) @ W[N,K](bf16)^T + bias(f32)
// Block tile 128x128, K-step 32; 8 waves (4 row x 2 col), wave tile 32x64
// (2x4 WMMA accumulators). Double-buffered LDS: DMA(k+1) overlaps WMMA(k).
// Requires M%128==0, N%128==0, K%32==0.
// flags bit0: (t,b)->(b,t) transposed store; bit1: store bf16 output.
// ---------------------------------------------------------------------------
#define BM  128
#define BN  128
#define BK  32
#define LDT 40   // LDS row stride in ushorts (32 data + 8 pad) -> 80B, 16B-aligned

__global__ __launch_bounds__(256) void gemm_bf16_kernel(
    const u16* __restrict__ A, const u16* __restrict__ W,
    const float* __restrict__ bias, void* __restrict__ Cout,
    int N, int K, int Tdim, int flags)
{
  __shared__ __align__(16) u16 Alds[2][BM * LDT];
  __shared__ __align__(16) u16 Blds[2][BN * LDT];

  const int tid     = threadIdx.x;
  const int lane    = tid & 31;
  const int wave    = tid >> 5;
  const int waveRow = wave >> 1;   // 0..3
  const int waveCol = wave & 1;    // 0..1
  const int lane15  = lane & 15;
  const int laneHi  = lane >> 4;
  const size_t aRow0 = (size_t)blockIdx.y * BM;
  const size_t bRow0 = (size_t)blockIdx.x * BN;

  // per-thread A staging: 2 x 16B chunks (row, quarter)
  const int row0 = tid >> 2,          q0 = tid & 3;
  const int row1 = (tid + 256) >> 2,  q1 = tid & 3;
  const u16* ga0 = A + (aRow0 + row0) * (size_t)K + q0 * 8;
  const u16* ga1 = A + (aRow0 + row1) * (size_t)K + q1 * 8;
  const unsigned la0 = (unsigned)(size_t)&Alds[0][row0 * LDT + q0 * 8];
  const unsigned la1 = (unsigned)(size_t)&Alds[0][row1 * LDT + q1 * 8];
  const unsigned lb  = (unsigned)(size_t)&Blds[0][0];
  const unsigned aBufStride = (unsigned)(BM * LDT * sizeof(u16));
  const unsigned bBufStride = (unsigned)(BN * LDT * sizeof(u16));
  const u16* gw = W + bRow0 * (size_t)K;

  const int nk = K / BK;

  v8f acc[2][4];
#pragma unroll
  for (int i = 0; i < 2; ++i)
#pragma unroll
    for (int j = 0; j < 4; ++j) acc[i][j] = (v8f){};

  auto issue = [&](int k) {
    unsigned buf = (unsigned)(k & 1);
    async_ld_b128(la0 + buf * aBufStride, ga0 + (size_t)k * BK);
    async_ld_b128(la1 + buf * aBufStride, ga1 + (size_t)k * BK);
    if (wave == 0)
      tdm_load_tile(lb + buf * bBufStride, gw + (size_t)k * BK, K);
  };

  issue(0);   // prologue: stage tile 0

  for (int k = 0; k < nk; ++k) {
    wait_async0();       // tile k's async A chunks landed (own ops only)
    wait_tensor0();      // tile k's TDM B tile landed (wave0; NOP elsewhere)
    __syncthreads();     // publish tile k; also: all waves done reading buf^1

    if (k + 1 < nk) issue(k + 1);   // DMA for k+1 overlaps WMMA for k

    const u16* Ab = Alds[k & 1];
    const u16* Bb = Blds[k & 1];

    // ---- fragments per ISA 7.12.2 (16-bit A 16x32 / B 32x16 layouts)
    FragU af[2], bf[4];
#pragma unroll
    for (int i = 0; i < 2; ++i) {
      const u16* p = &Ab[(waveRow * 32 + i * 16 + lane15) * LDT + laneHi * 8];
      af[i].q[0] = *(const uint4*)p;
      af[i].q[1] = *(const uint4*)(p + 16);
    }
#pragma unroll
    for (int j = 0; j < 4; ++j) {
      const u16* p = &Bb[(waveCol * 64 + j * 16 + lane15) * LDT + laneHi * 8];
      bf[j].q[0] = *(const uint4*)p;
      bf[j].q[1] = *(const uint4*)(p + 16);
    }
#pragma unroll
    for (int i = 0; i < 2; ++i)
#pragma unroll
      for (int j = 0; j < 4; ++j)
        acc[i][j] = __builtin_amdgcn_wmma_f32_16x16x32_bf16(
            false, af[i].v, false, bf[j].v, (short)0, acc[i][j], false, false);
  }

  // ---- epilogue: C/D layout -> lanes0-15 M=r, lanes16-31 M=r+8
  const int colBase = (int)bRow0 + waveCol * 64 + lane15;
  const int rowBase = (int)aRow0 + waveRow * 32 + laneHi * 8;
#pragma unroll
  for (int i = 0; i < 2; ++i) {
#pragma unroll
    for (int j = 0; j < 4; ++j) {
      int col = colBase + j * 16;
      float bv = bias ? bias[col] : 0.0f;
#pragma unroll
      for (int r = 0; r < 8; ++r) {
        size_t row = (size_t)rowBase + i * 16 + r;
        float val = acc[i][j][r] + bv;
        size_t o = (flags & 1)
            ? (((row & 1023) * (size_t)Tdim + (row >> 10)) * (size_t)N + col)
            : (row * (size_t)N + col);
        if (flags & 2) ((u16*)Cout)[o] = (u16)f2bf(val);
        else           ((float*)Cout)[o] = val;
      }
    }
  }
}

// ---------------------------------------------------------------------------
// Elementwise helpers
// ---------------------------------------------------------------------------
__global__ __launch_bounds__(256) void zero_kernel(float* __restrict__ p, int n) {
  int i = blockIdx.x * 256 + threadIdx.x;
  if (i < n) p[i] = 0.0f;
}

__global__ __launch_bounds__(256) void cvt_bf16_kernel(
    const float* __restrict__ a, u16* __restrict__ o, long n4)
{
  long i = (long)blockIdx.x * 256 + threadIdx.x;
  if (i >= n4) return;
  float4 v = ((const float4*)a)[i];
  ((uint2*)o)[i] = make_uint2(f2bf(v.x) | (f2bf(v.y) << 16),
                              f2bf(v.z) | (f2bf(v.w) << 16));
}
__global__ __launch_bounds__(256) void cvt2_bf16_kernel(
    const float* __restrict__ a, const float* __restrict__ b,
    u16* __restrict__ o, long n4)
{
  long i = (long)blockIdx.x * 256 + threadIdx.x;
  if (i >= n4) return;
  float4 va = ((const float4*)a)[i];
  float4 vb = ((const float4*)b)[i];
  va.x += vb.x; va.y += vb.y; va.z += vb.z; va.w += vb.w;
  ((uint2*)o)[i] = make_uint2(f2bf(va.x) | (f2bf(va.y) << 16),
                              f2bf(va.z) | (f2bf(va.w) << 16));
}

// h = (1-z)*n + z*h ; also emits bf16 copy of the new h
__global__ __launch_bounds__(256) void gru_fuse_kernel(
    const float* __restrict__ gi, const float* __restrict__ gh,
    float* __restrict__ h, u16* __restrict__ h_bf)
{
  int idx = blockIdx.x * 256 + threadIdx.x;
  int b = idx >> 10, j = idx & 1023;
  size_t base = (size_t)b * (3 * H_DIM) + j;
  float ir = gi[base],             hr = gh[base];
  float iz = gi[base + H_DIM],     hz = gh[base + H_DIM];
  float in = gi[base + 2 * H_DIM], hn = gh[base + 2 * H_DIM];
  float r = 1.0f / (1.0f + __expf(-(ir + hr)));
  float z = 1.0f / (1.0f + __expf(-(iz + hz)));
  float n = tanhf(in + r * hn);
  float hv = (1.0f - z) * n + z * h[idx];
  h[idx] = hv;
  h_bf[idx] = (u16)f2bf(hv);
}

// out = LayerNorm(a + b) * w + beta (+ optional bf16 copy), block per row
__global__ __launch_bounds__(256) void add_layernorm_kernel(
    const float* __restrict__ a, const float* __restrict__ b,
    const float* __restrict__ w, const float* __restrict__ bet,
    float* __restrict__ out, u16* __restrict__ obf)
{
  __shared__ float red[256];
  int row = blockIdx.x, tid = threadIdx.x;
  const float* pa = a + (size_t)row * H_DIM;
  const float* pb = b + (size_t)row * H_DIM;
  float v[4]; float s = 0.0f;
#pragma unroll
  for (int i = 0; i < 4; ++i) { v[i] = pa[tid + i * 256] + pb[tid + i * 256]; s += v[i]; }
  red[tid] = s; __syncthreads();
  for (int o = 128; o > 0; o >>= 1) { if (tid < o) red[tid] += red[tid + o]; __syncthreads(); }
  float mu = red[0] * (1.0f / H_DIM);
  __syncthreads();
  s = 0.0f;
#pragma unroll
  for (int i = 0; i < 4; ++i) { float d = v[i] - mu; s += d * d; }
  red[tid] = s; __syncthreads();
  for (int o = 128; o > 0; o >>= 1) { if (tid < o) red[tid] += red[tid + o]; __syncthreads(); }
  float inv = rsqrtf(red[0] * (1.0f / H_DIM) + 1e-5f);
#pragma unroll
  for (int i = 0; i < 4; ++i) {
    int j = tid + i * 256;
    float val = (v[i] - mu) * inv * w[j] + bet[j];
    out[(size_t)row * H_DIM + j] = val;
    if (obf) obf[(size_t)row * H_DIM + j] = (u16)f2bf(val);
  }
}

// ---------------------------------------------------------------------------
// Encoder attention: block per batch element, 8 heads in pairs.
// ctx (bf16) = softmax(qK^T/sqrt(128)) V ; aws (f32) = head-mean attention.
// ---------------------------------------------------------------------------
__global__ __launch_bounds__(256) void enc_attn_kernel(
    const float* __restrict__ q, const float* __restrict__ Km,
    const float* __restrict__ Vm, u16* __restrict__ ctx,
    float* __restrict__ aws)
{
  __shared__ float qsh[256];
  __shared__ float sc[256];
  __shared__ float esc[256];
  __shared__ float awsh[SEQ_N];

  const int b = blockIdx.x, tid = threadIdx.x;
  const int hh = tid >> 7;
  const int s  = tid & 127;
  if (tid < SEQ_N) awsh[tid] = 0.0f;

  for (int it = 0; it < 4; ++it) {
    __syncthreads();
    qsh[tid] = q[(size_t)b * H_DIM + it * 256 + tid];
    __syncthreads();

    const float* kp = Km + ((size_t)b * SEQ_N + s) * H_DIM + it * 256 + hh * 128;
    const float* qp = &qsh[hh * 128];
    float acc = 0.0f;
#pragma unroll 4
    for (int d = 0; d < 128; ++d) acc += qp[d] * kp[d];
    sc[tid] = acc * 0.08838834764831845f;
    __syncthreads();

    const float* seg = &sc[hh * 128];
    float mx = -3.4e38f;
    for (int j = 0; j < 128; ++j) mx = fmaxf(mx, seg[j]);
    float e = __expf(sc[tid] - mx);
    esc[tid] = e;
    __syncthreads();
    const float* eseg = &esc[hh * 128];
    float sum = 0.0f;
    for (int j = 0; j < 128; ++j) sum += eseg[j];
    float attn = e / sum;
    __syncthreads();
    esc[tid] = attn;
    __syncthreads();
    if (tid < SEQ_N) awsh[tid] += (esc[tid] + esc[128 + tid]) * 0.125f;

    const float* vp = Vm + (size_t)b * SEQ_N * H_DIM + it * 256 + hh * 128 + s;
    const float* aseg = &esc[hh * 128];
    float c = 0.0f;
#pragma unroll 4
    for (int s2 = 0; s2 < 128; ++s2) c += aseg[s2] * vp[(size_t)s2 * H_DIM];
    ctx[(size_t)b * H_DIM + it * 256 + hh * 128 + s] = (u16)f2bf(c);
  }
  __syncthreads();
  if (tid < SEQ_N) aws[(size_t)b * SEQ_N + tid] = awsh[tid];
}

// ---------------------------------------------------------------------------
// Orchestration
// ---------------------------------------------------------------------------
extern "C" void kernel_launch(void* const* d_in, const int* in_sizes, int n_in,
                              void* d_out, int out_size, void* d_ws, size_t ws_size,
                              hipStream_t stream)
{
  (void)in_sizes; (void)n_in; (void)out_size; (void)ws_size;

  const float* decoder_input   = (const float*)d_in[0];
  const float* encoder_outputs = (const float*)d_in[1];
  const float* w_ih   = (const float*)d_in[2];
  const float* w_hh   = (const float*)d_in[3];
  const float* b_ih   = (const float*)d_in[4];
  const float* b_hh   = (const float*)d_in[5];
  const float* enc_w  = (const float*)d_in[6];
  const float* enc_b  = (const float*)d_in[7];
  const float* sa_in_w  = (const float*)d_in[8];
  const float* sa_in_b  = (const float*)d_in[9];
  const float* sa_out_w = (const float*)d_in[10];
  const float* sa_out_b = (const float*)d_in[11];
  const float* sa_fc_w  = (const float*)d_in[12];
  const float* sa_fc_b  = (const float*)d_in[13];
  const float* ln_w     = (const float*)d_in[14];
  const float* ln_b     = (const float*)d_in[15];
  const float* ea_in_w  = (const float*)d_in[16];
  const float* ea_in_b  = (const float*)d_in[17];
  const float* ea_out_w = (const float*)d_in[18];
  const float* ea_out_b = (const float*)d_in[19];
  const float* fc_w     = (const float*)d_in[20];
  const float* fc_b     = (const float*)d_in[21];
  const float* op_w     = (const float*)d_in[22];
  const float* op_b     = (const float*)d_in[23];

  const size_t MS  = (size_t)BATCH_N * SEQ_N;   // 131072
  const size_t HH  = (size_t)H_DIM * H_DIM;     // 1M
  const size_t BH  = (size_t)BATCH_N * H_DIM;   // 1M

  // byte-based workspace allocator (256B aligned)
  char* wsb = (char*)d_ws;
  size_t off = 0;
  auto alloc = [&](size_t bytes) -> void* {
    void* p = wsb + off;
    off += (bytes + 255) & ~(size_t)255;
    return p;
  };
  u16*   encin_bf = (u16*)alloc(MS * H_DIM * 2);
  u16*   enc_bf   = (u16*)alloc(MS * H_DIM * 2);
  float* Km       = (float*)alloc(MS * H_DIM * 4);
  float* Vm       = (float*)alloc(MS * H_DIM * 4);
  float* h        = (float*)alloc(BH * 4);
  u16*   h_bf     = (u16*)alloc(BH * 2);
  float* gi       = (float*)alloc(BH * 3 * 4);
  float* gh       = (float*)alloc(BH * 3 * 4);
  u16*   tv1_bf   = (u16*)alloc(BH * 2);
  float* tsa      = (float*)alloc(BH * 4);
  float* txln     = (float*)alloc(BH * 4);
  u16*   txln_bf  = (u16*)alloc(BH * 2);
  float* txf      = (float*)alloc(BH * 4);
  float* tq       = (float*)alloc(BH * 4);
  u16*   tctxi_bf = (u16*)alloc(BH * 2);
  float* tctx     = (float*)alloc(BH * 4);
  u16*   hc_bf    = (u16*)alloc(BH * 2);
  u16*   dec0_bf  = (u16*)alloc(BH * 2);
  u16*   outs_bf  = (u16*)alloc((size_t)TSTEPS * BH * 2);
  // bf16 weights
  u16* w_ih_bf   = (u16*)alloc(3 * HH * 2);
  u16* w_hh_bf   = (u16*)alloc(3 * HH * 2);
  u16* ea_in_bf  = (u16*)alloc(3 * HH * 2);
  u16* sa_v_bf   = (u16*)alloc(HH * 2);
  u16* sa_out_bf = (u16*)alloc(HH * 2);
  u16* sa_fc_bf  = (u16*)alloc(HH * 2);
  u16* ea_out_bf = (u16*)alloc(HH * 2);
  u16* fc_bf     = (u16*)alloc(HH * 2);
  u16* op_bf     = (u16*)alloc(HH * 2);
  u16* enc_w_bf  = (u16*)alloc(HH * 2);

  float* out_f    = (float*)d_out;
  float* aws_base = out_f + (size_t)BATCH_N * TSTEPS * H_DIM;

  auto cvt = [&](const float* a, u16* o, size_t n) {
    long n4 = (long)(n / 4);
    cvt_bf16_kernel<<<(unsigned)((n4 + 255) / 256), 256, 0, stream>>>(a, o, n4);
  };
  auto cvt2 = [&](const float* a, const float* b, u16* o, size_t n) {
    long n4 = (long)(n / 4);
    cvt2_bf16_kernel<<<(unsigned)((n4 + 255) / 256), 256, 0, stream>>>(a, b, o, n4);
  };
  auto gemm = [&](const u16* A, const u16* Wp, const float* bp, void* Cp,
                  int M, int N, int K, int Tdim, int flags) {
    dim3 grid((unsigned)(N / BN), (unsigned)(M / BM));
    gemm_bf16_kernel<<<grid, 256, 0, stream>>>(A, Wp, bp, Cp, N, K, Tdim, flags);
  };

  // ---- one-time conversions -------------------------------------------------
  zero_kernel<<<(BATCH_N * H_DIM) / 256, 256, 0, stream>>>(h, BATCH_N * H_DIM);
  cvt(h, h_bf, BH);                                  // h0 = 0 in bf16 too
  cvt(decoder_input, dec0_bf, BH);
  cvt(encoder_outputs, encin_bf, MS * H_DIM);
  cvt(w_ih, w_ih_bf, 3 * HH);
  cvt(w_hh, w_hh_bf, 3 * HH);
  cvt(ea_in_w, ea_in_bf, 3 * HH);
  cvt(sa_in_w + 2 * HH, sa_v_bf, HH);
  cvt(sa_out_w, sa_out_bf, HH);
  cvt(sa_fc_w, sa_fc_bf, HH);
  cvt(ea_out_w, ea_out_bf, HH);
  cvt(fc_w, fc_bf, HH);
  cvt(op_w, op_bf, HH);
  cvt(enc_w, enc_w_bf, HH);

  // ---- hoisted encoder projection + loop-invariant K/V ----------------------
  gemm(encin_bf, enc_w_bf, enc_b, enc_bf, (int)MS, H_DIM, H_DIM, 1, /*bf out*/2);
  gemm(enc_bf, ea_in_bf + 1 * HH, ea_in_b + 1 * H_DIM, Km, (int)MS, H_DIM, H_DIM, 1, 0);
  gemm(enc_bf, ea_in_bf + 2 * HH, ea_in_b + 2 * H_DIM, Vm, (int)MS, H_DIM, H_DIM, 1, 0);

  const u16* dec_bf = dec0_bf;
  for (int t = 0; t < TSTEPS; ++t) {
    // GRU cell
    gemm(dec_bf, w_ih_bf, b_ih, gi, BATCH_N, 3 * H_DIM, H_DIM, 1, 0);
    gemm(h_bf,   w_hh_bf, b_hh, gh, BATCH_N, 3 * H_DIM, H_DIM, 1, 0);
    gru_fuse_kernel<<<(BATCH_N * H_DIM) / 256, 256, 0, stream>>>(gi, gh, h, h_bf);

    // Self-attn block (length-1 softmax == identity -> V path only)
    gemm(h_bf, sa_v_bf, sa_in_b + 2 * H_DIM, tv1_bf, BATCH_N, H_DIM, H_DIM, 1, 2);
    gemm(tv1_bf, sa_out_bf, sa_out_b, tsa, BATCH_N, H_DIM, H_DIM, 1, 0);
    add_layernorm_kernel<<<BATCH_N, 256, 0, stream>>>(tsa, h, ln_w, ln_b, txln, txln_bf);
    gemm(txln_bf, sa_fc_bf, sa_fc_b, txf, BATCH_N, H_DIM, H_DIM, 1, 0);
    add_layernorm_kernel<<<BATCH_N, 256, 0, stream>>>(txf, h, ln_w, ln_b, h, h_bf);

    // Encoder attention (K/V precomputed)
    gemm(h_bf, ea_in_bf, ea_in_b, tq, BATCH_N, H_DIM, H_DIM, 1, 0);
    enc_attn_kernel<<<BATCH_N, 256, 0, stream>>>(tq, Km, Vm, tctxi_bf,
        aws_base + (size_t)t * BATCH_N * SEQ_N);
    gemm(tctxi_bf, ea_out_bf, ea_out_b, tctx, BATCH_N, H_DIM, H_DIM, 1, 0);

    // out_t = (h + ctx) @ fc_w^T + fc_b, stored directly as bf16 (next dec)
    cvt2(h, tctx, hc_bf, BH);
    u16* out_t = outs_bf + (size_t)t * BH;
    gemm(hc_bf, fc_bf, fc_b, out_t, BATCH_N, H_DIM, H_DIM, 1, 2);
    dec_bf = out_t;
  }

  // Final projection, (t,b) -> (b,t) transpose fused into the epilogue.
  gemm(outs_bf, op_bf, op_b, out_f, TSTEPS * BATCH_N, H_DIM, H_DIM, TSTEPS, 1);
}
